// GPT_10720238371231
// MI455X (gfx1250) — matrix operand order repaired and checked by reading, not compiled, not run on hardware
//
#include <hip/hip_runtime.h>
#include <hip/hip_bf16.h>
#include <math.h>

typedef __attribute__((ext_vector_type(16))) _Float16 v16h;
typedef __attribute__((ext_vector_type(2)))  _Float16 h2;
typedef __attribute__((ext_vector_type(8)))  float    v8f;

#define L_  8
#define D_  1024
#define H_  16
#define V_  32000
#define T_  1024
#define B_  2
#define HD_ 64
#define BT_ (B_*T_)
#define EPSF 1e-5f

// ---------------------------------------------------------------------------
// Embedding: x[b,t,:] = tok_emb[idx[b,t],:] + pos_emb[0,t,:]
// ---------------------------------------------------------------------------
__global__ void embed_kernel(const int* __restrict__ idx,
                             const float* __restrict__ tok,
                             const float* __restrict__ pos,
                             float* __restrict__ x) {
  int gid = blockIdx.x * blockDim.x + threadIdx.x;
  if (gid >= BT_ * D_) return;
  int row = gid / D_;            // b*T + t
  int d   = gid - row * D_;
  int t   = row % T_;
  x[gid] = tok[(size_t)idx[row] * D_ + d] + pos[(size_t)t * D_ + d];
}

// ---------------------------------------------------------------------------
// LayerNorm over last dim (D=1024), one block of 256 threads per row
// ---------------------------------------------------------------------------
__global__ void ln_kernel(const float* __restrict__ x,
                          const float* __restrict__ s,
                          const float* __restrict__ b,
                          float* __restrict__ out) {
  __shared__ float red[256];
  int row = blockIdx.x;
  const float* xr = x + (size_t)row * D_;
  float sum = 0.f, sq = 0.f;
  for (int i = threadIdx.x; i < D_; i += 256) { float v = xr[i]; sum += v; sq += v * v; }
  red[threadIdx.x] = sum; __syncthreads();
  for (int o = 128; o > 0; o >>= 1) {
    if (threadIdx.x < o) red[threadIdx.x] += red[threadIdx.x + o];
    __syncthreads();
  }
  float mean = red[0] / (float)D_;
  __syncthreads();
  red[threadIdx.x] = sq; __syncthreads();
  for (int o = 128; o > 0; o >>= 1) {
    if (threadIdx.x < o) red[threadIdx.x] += red[threadIdx.x + o];
    __syncthreads();
  }
  float var = red[0] / (float)D_ - mean * mean;
  float inv = rsqrtf(var + EPSF);
  for (int i = threadIdx.x; i < D_; i += 256)
    out[(size_t)row * D_ + i] = (xr[i] - mean) * inv * s[i] + b[i];
}

// ---------------------------------------------------------------------------
// WMMA GEMM: C[M,N] = A[M,K] @ W[K,N] (+bias) (+residual) (+GELU)
//
// Block = 256 threads (8 waves), tile 128x64, BK=32. fp32 -> f16 conversion
// happens while staging into LDS, and tiles are stored in *fragment-major*
// layout: the 16 halves each lane needs for v_wmma are contiguous 32 bytes,
// so fragment loads are 2x ds_load_b128 instead of 16x ds_load_u16.
//
// Fragment slot inversion (from ISA 16-bit A/B VGPR layouts):
//   A (16x32): half = (k>>3)&1, e = (k&7) + (k>=16 ? 8 : 0), lane = half*16+m
//   B (32x16): half = (k>>3)&1, e = (k&7) + (k>=16 ? 8 : 0), lane = half*16+n
// Even-k pairs map to adjacent e,e+1 -> one packed ds_store_b32 per pair.
// ---------------------------------------------------------------------------
#define BM 128
#define BN 64
#define BK 32

__global__ __launch_bounds__(256) void gemm_wmma_kernel(
    const float* __restrict__ A, const float* __restrict__ W,
    const float* __restrict__ bias, const float* __restrict__ res,
    float* __restrict__ C, int M, int N, int K, int epi) {
  __shared__ _Float16 As[8 * 32 * 16];   // 8 row-groups x 32 lanes x 16 halves
  __shared__ _Float16 Bs[4 * 32 * 16];   // 4 col-groups x 32 lanes x 16 halves
  int tid  = threadIdx.x;                // 0..255
  int w    = tid >> 5;                   // wave: rows [w*16, w*16+16)
  int lane = tid & 31;
  int half = lane >> 4;
  int l15  = lane & 15;
  int m0 = blockIdx.y * BM;
  int n0 = blockIdx.x * BN;

  v8f acc0 = {}, acc1 = {}, acc2 = {}, acc3 = {};

  for (int k0 = 0; k0 < K; k0 += BK) {
    // Prefetch next K-tile of A and W toward L2/WGP$ (global_prefetch_b8)
    if (k0 + BK < K) {
      __builtin_prefetch(&A[(size_t)(m0 + (tid >> 1)) * K + k0 + BK + (tid & 1) * 16], 0, 3);
      if (tid < 128)
        __builtin_prefetch(&W[(size_t)(k0 + BK + (tid >> 2)) * N + n0 + (tid & 3) * 16], 0, 3);
    }

    // Stage A tile (128x32): 2048 even-k pairs, 8 per thread
#pragma unroll
    for (int i = 0; i < 8; ++i) {
      int p   = i * 256 + tid;
      int row = p >> 4;            // 0..127
      int kp  = p & 15;            // pair index, k = 2*kp
      int k   = kp * 2;
      const float* src = &A[(size_t)(m0 + row) * K + (k0 + k)];
      h2 pk; pk[0] = (_Float16)src[0]; pk[1] = (_Float16)src[1];
      int hf = (kp >> 2) & 1;                       // (k>>3)&1
      int e  = (k & 7) + ((k >= 16) ? 8 : 0);
      int g  = row >> 4;
      int ln = hf * 16 + (row & 15);
      *(h2*)&As[((g * 32 + ln) << 4) + e] = pk;
    }
    // Stage B tile (32x64): 1024 even-k pairs, 4 per thread
#pragma unroll
    for (int i = 0; i < 4; ++i) {
      int p  = i * 256 + tid;
      int n  = p & 63;
      int kp = p >> 6;             // pair index, k = 2*kp
      int k  = kp * 2;
      const float* src = &W[(size_t)(k0 + k) * N + (n0 + n)];
      h2 pk; pk[0] = (_Float16)src[0]; pk[1] = (_Float16)src[N];
      int hf = (kp >> 2) & 1;                       // (k>>3)&1
      int e  = (k & 7) + ((k >= 16) ? 8 : 0);
      int g  = n >> 4;
      int ln = hf * 16 + (n & 15);
      *(h2*)&Bs[((g * 32 + ln) << 4) + e] = pk;
    }
    __syncthreads();

    // Fragment loads: contiguous 32B per lane -> wide ds loads
    v16h a = *(const v16h*)&As[(w * 32 + lane) << 4];
    v16h b0f = *(const v16h*)&Bs[(0 * 32 + lane) << 4];
    v16h b1f = *(const v16h*)&Bs[(1 * 32 + lane) << 4];
    v16h b2f = *(const v16h*)&Bs[(2 * 32 + lane) << 4];
    v16h b3f = *(const v16h*)&Bs[(3 * 32 + lane) << 4];

    acc0 = __builtin_amdgcn_wmma_f32_16x16x32_f16(false, a, false, b0f, (short)0, acc0, false, false);
    acc1 = __builtin_amdgcn_wmma_f32_16x16x32_f16(false, a, false, b1f, (short)0, acc1, false, false);
    acc2 = __builtin_amdgcn_wmma_f32_16x16x32_f16(false, a, false, b2f, (short)0, acc2, false, false);
    acc3 = __builtin_amdgcn_wmma_f32_16x16x32_f16(false, a, false, b3f, (short)0, acc3, false, false);
    __syncthreads();
  }

  // Store: C/D layout — VGPR r holds row r + 8*half, col = lane&15
#pragma unroll
  for (int g = 0; g < 4; ++g) {
    v8f acc = (g == 0) ? acc0 : (g == 1) ? acc1 : (g == 2) ? acc2 : acc3;
    int col = n0 + g * 16 + l15;
#pragma unroll
    for (int r = 0; r < 8; ++r) {
      int row = m0 + w * 16 + r + 8 * half;
      float v = acc[r];
      if (bias) v += bias[col];
      if (res)  v += res[(size_t)row * N + col];
      if (epi == 1) v = 0.5f * v * (1.0f + erff(v * 0.70710678118654752f));
      C[(size_t)row * N + col] = v;
    }
  }
}

// ---------------------------------------------------------------------------
// Flash-style attention, one wave32 per (b,h,query). HD=64 -> 2 dims/lane.
// qkv layout per row: [Q(D) | K(D) | V(D)], head h at offset h*HD.
// Online softmax over T=1024 keys in chunks of 32 (one key per lane).
// ---------------------------------------------------------------------------
__global__ void attn_kernel(const float* __restrict__ qkv, float* __restrict__ y) {
  __shared__ float qsh[8 * 64];
  int wli  = threadIdx.x >> 5;
  int lane = threadIdx.x & 31;
  int wid  = blockIdx.x * 8 + wli;        // < B*H*T (exact)
  int qt = wid % T_;
  int h  = (wid / T_) % H_;
  int b  = wid / (T_ * H_);

  const float* base  = qkv + (size_t)b * T_ * 3 * D_;
  const float* Qrow  = base + (size_t)qt * 3 * D_ + h * HD_;
  const float* Kbase = base + D_ + h * HD_;
  const float* Vbase = base + 2 * D_ + h * HD_;

  float* qs = qsh + wli * 64;
  qs[2 * lane]     = Qrow[2 * lane];
  qs[2 * lane + 1] = Qrow[2 * lane + 1];
  __syncthreads();

  float m = -1e30f, lsum = 0.f, a0 = 0.f, a1 = 0.f;
  for (int kb = 0; kb < T_; kb += 32) {
    const float* Krow = Kbase + (size_t)(kb + lane) * 3 * D_;
    float s = 0.f;
#pragma unroll
    for (int d = 0; d < HD_; ++d) s += qs[d] * Krow[d];
    s *= 0.125f;                          // 1/sqrt(64)
    float cm = s;
#pragma unroll
    for (int o = 16; o > 0; o >>= 1) cm = fmaxf(cm, __shfl_xor(cm, o, 32));
    float mnew  = fmaxf(m, cm);
    float scale = __expf(m - mnew);
    float p     = __expf(s - mnew);
    float ps = p;
#pragma unroll
    for (int o = 16; o > 0; o >>= 1) ps += __shfl_xor(ps, o, 32);
    lsum = lsum * scale + ps;
    a0 *= scale; a1 *= scale;
    for (int j = 0; j < 32; ++j) {
      float pj = __shfl(p, j, 32);
      const float* Vrow = Vbase + (size_t)(kb + j) * 3 * D_;
      a0 += pj * Vrow[2 * lane];
      a1 += pj * Vrow[2 * lane + 1];
    }
    m = mnew;
  }
  float invl = 1.0f / lsum;
  float* yr = y + (size_t)(b * T_ + qt) * D_ + h * HD_;
  yr[2 * lane]     = a0 * invl;
  yr[2 * lane + 1] = a1 * invl;
}

// ---------------------------------------------------------------------------
// Loss = mean over tokens of -(logit[target] - max - log(sum exp))
// ---------------------------------------------------------------------------
__global__ void zero_loss_kernel(float* loss) {
  if (threadIdx.x == 0 && blockIdx.x == 0) loss[0] = 0.f;
}

__global__ void loss_kernel(const float* __restrict__ logits,
                            const int* __restrict__ targets,
                            float* __restrict__ loss) {
  __shared__ float red[256];
  int row = blockIdx.x;
  const float* lr = logits + (size_t)row * V_;
  float mx = -1e30f;
  for (int i = threadIdx.x; i < V_; i += 256) mx = fmaxf(mx, lr[i]);
  red[threadIdx.x] = mx; __syncthreads();
  for (int o = 128; o > 0; o >>= 1) {
    if (threadIdx.x < o) red[threadIdx.x] = fmaxf(red[threadIdx.x], red[threadIdx.x + o]);
    __syncthreads();
  }
  mx = red[0];
  __syncthreads();
  float se = 0.f;
  for (int i = threadIdx.x; i < V_; i += 256) se += __expf(lr[i] - mx);
  red[threadIdx.x] = se; __syncthreads();
  for (int o = 128; o > 0; o >>= 1) {
    if (threadIdx.x < o) red[threadIdx.x] += red[threadIdx.x + o];
    __syncthreads();
  }
  if (threadIdx.x == 0) {
    float nll = logf(red[0]) + mx - lr[targets[row]];
    atomicAdd(loss, nll * (1.0f / (float)BT_));
  }
}

// ---------------------------------------------------------------------------
// Host-side orchestration (all on `stream`, graph-capture safe)
// ---------------------------------------------------------------------------
extern "C" void kernel_launch(void* const* d_in, const int* in_sizes, int n_in,
                              void* d_out, int out_size, void* d_ws, size_t ws_size,
                              hipStream_t stream) {
  (void)in_sizes; (void)n_in; (void)out_size; (void)ws_size;

  const int*   idx     = (const int*)d_in[0];
  const int*   targets = (const int*)d_in[1];
  const float* tok_emb = (const float*)d_in[2];
  const float* pos_emb = (const float*)d_in[3];
  const float* ln1_s   = (const float*)d_in[4];
  const float* ln1_b   = (const float*)d_in[5];
  const float* qkv_w   = (const float*)d_in[6];
  const float* qkv_b   = (const float*)d_in[7];
  const float* proj_w  = (const float*)d_in[8];
  const float* proj_b  = (const float*)d_in[9];
  const float* ln2_s   = (const float*)d_in[10];
  const float* ln2_b   = (const float*)d_in[11];
  const float* fc1_w   = (const float*)d_in[12];
  const float* fc1_b   = (const float*)d_in[13];
  const float* fc2_w   = (const float*)d_in[14];
  const float* fc2_b   = (const float*)d_in[15];
  const float* lnf_s   = (const float*)d_in[16];
  const float* lnf_b   = (const float*)d_in[17];
  const float* head_w  = (const float*)d_in[18];

  float* logits = (float*)d_out;                    // [BT, V]
  float* loss   = logits + (size_t)BT_ * V_;        // 1 float

  float* ws   = (float*)d_ws;
  float* x    = ws;                                 // [BT, D]
  float* hb   = x    + (size_t)BT_ * D_;            // [BT, D]
  float* qkvb = hb   + (size_t)BT_ * D_;            // [BT, 3D]
  float* yb   = qkvb + (size_t)BT_ * 3 * D_;        // [BT, D]
  float* f1   = yb   + (size_t)BT_ * D_;            // [BT, 4D]

  embed_kernel<<<(BT_ * D_ + 255) / 256, 256, 0, stream>>>(idx, tok_emb, pos_emb, x);

  for (int l = 0; l < L_; ++l) {
    ln_kernel<<<BT_, 256, 0, stream>>>(x, ln1_s + (size_t)l * D_, ln1_b + (size_t)l * D_, hb);
    gemm_wmma_kernel<<<dim3(3 * D_ / BN, BT_ / BM), 256, 0, stream>>>(
        hb, qkv_w + (size_t)l * D_ * 3 * D_, qkv_b + (size_t)l * 3 * D_,
        nullptr, qkvb, BT_, 3 * D_, D_, 0);
    attn_kernel<<<(B_ * H_ * T_) / 8, 256, 0, stream>>>(qkvb, yb);
    gemm_wmma_kernel<<<dim3(D_ / BN, BT_ / BM), 256, 0, stream>>>(
        yb, proj_w + (size_t)l * D_ * D_, proj_b + (size_t)l * D_,
        x, x, BT_, D_, D_, 0);
    ln_kernel<<<BT_, 256, 0, stream>>>(x, ln2_s + (size_t)l * D_, ln2_b + (size_t)l * D_, hb);
    gemm_wmma_kernel<<<dim3(4 * D_ / BN, BT_ / BM), 256, 0, stream>>>(
        hb, fc1_w + (size_t)l * D_ * 4 * D_, fc1_b + (size_t)l * 4 * D_,
        nullptr, f1, BT_, 4 * D_, D_, 1);
    gemm_wmma_kernel<<<dim3(D_ / BN, BT_ / BM), 256, 0, stream>>>(
        f1, fc2_w + (size_t)l * 4 * D_ * D_, fc2_b + (size_t)l * D_,
        x, x, BT_, D_, 4 * D_, 0);
  }

  ln_kernel<<<BT_, 256, 0, stream>>>(x, lnf_s, lnf_b, hb);
  gemm_wmma_kernel<<<dim3(V_ / BN, BT_ / BM), 256, 0, stream>>>(
      hb, head_w, nullptr, nullptr, logits, BT_, V_, D_, 0);

  zero_loss_kernel<<<1, 32, 0, stream>>>(loss);
  loss_kernel<<<BT_, 256, 0, stream>>>(logits, targets, loss);
}
